// InputDrivenConvolutionDecider_3770981285924
// MI455X (gfx1250) — compile-verified, matrix operand order checked by reading, not run on hardware
//
#include <hip/hip_runtime.h>

// ---------------------------------------------------------------------------
// MoE inception-conv + pool + dense for MI455X (gfx1250), wave32, WMMA bf16.
//
// N=2048 samples, 64x64x8 images, E=4 experts x G=512, out 2048x512 fp32.
//
// Roofline: 268 MB input @ 23.3 TB/s => ~11.5 us floor; compute via bf16 WMMA
// kept well under that. The linear 1x1 skip is pooled analytically in fp32
// (mean(x) @ Wskip), so WMMA only evaluates the two relu branches:
//   fused conv GEMM: M=16 pixels, K=96 (3x3x8 padded from 72), N=16 cols:
//     cols 0..7  : pre-relu conv1x1 (center tap rows only)
//     cols 8..15 : pre-relu conv3x3
// Image staged to LDS as bf16 with a zeroed 1-px halo (66x66x8) so all tap
// reads are unconditional ds_load_b128 (no exec-mask branching in the loop).
// ---------------------------------------------------------------------------

typedef __attribute__((ext_vector_type(16))) __bf16 v16bf;
typedef __attribute__((ext_vector_type(8)))  __bf16 v8bf;
typedef __attribute__((ext_vector_type(4)))  __bf16 v4bf;
typedef __attribute__((ext_vector_type(8)))  float  v8f;

#define NSAMP     2048
#define HW        64
#define HP        66               // padded dim (1-px halo)
#define CH        8
#define EXPERTS   4
#define MOUT      512
#define PIX       (HW * HW)        // 4096
#define IMG_ELEMS (PIX * CH)       // 32768 floats per sample
#define FRAG_PER_E (3 * 32 * 16)   // ksteps * lanes * elems = 1536

// ---------------------------------------------------------------------------
// Pre-pack fused conv weights into per-lane WMMA B-fragment layout (bf16).
// Mirrors the documented 16-bit A layout:
//   lane half b = (lane>=16)?8:0 ; elem i<8  -> K = 32*ks + b + i
//                                  elem i>=8 -> K = 32*ks + 16 + b + (i-8)
//   column n = lane & 15
// K index kk in [0,72): tap t = kk>>3 (ty=t/3, tx=t%3), channel c = kk&7.
// ---------------------------------------------------------------------------
__global__ __launch_bounds__(256)
void prep_bfrag_kernel(const float* __restrict__ W1,   // [E,1,1,8,8]
                       const float* __restrict__ W3,   // [E,3,3,8,8]
                       __bf16* __restrict__ frag)      // [E,3,32,16]
{
    int gid = blockIdx.x * 256 + threadIdx.x;
    if (gid >= EXPERTS * FRAG_PER_E) return;

    int e    = gid / FRAG_PER_E;
    int r    = gid % FRAG_PER_E;
    int ks   = r / 512;
    int r2   = r % 512;
    int lane = r2 / 16;
    int i    = r2 & 15;

    int b  = (lane >= 16) ? 8 : 0;
    int kk = 32 * ks + b + ((i < 8) ? i : (16 + (i - 8)));
    int col = lane & 15;

    float v = 0.0f;
    if (kk < 72) {
        int t = kk >> 3, c = kk & 7;
        if (col < 8) {                       // conv1x1 branch: center tap only
            if (t == 4) v = W1[e * 64 + c * 8 + col];
        } else {                             // conv3x3 branch: all 9 taps
            int ty = t / 3, tx = t % 3;
            v = W3[((e * 3 + ty) * 3 + tx) * 64 + c * 8 + (col - 8)];
        }
    }
    frag[gid] = (__bf16)v;
}

// ---------------------------------------------------------------------------
// Main kernel: one workgroup (256 threads = 8 wave32) per sample.
// ---------------------------------------------------------------------------
__global__ __launch_bounds__(256)
void moe_conv_pool_dense_kernel(const float* __restrict__ inputs,   // [N,64,64,8]
                                const int*   __restrict__ indices,  // [E*G]
                                const float* __restrict__ Wskip,    // [E,1,1,8,16]
                                const float* __restrict__ Wd,       // [E,16,512]
                                const float* __restrict__ cmask,    // [E,512]
                                const __bf16* __restrict__ bfrag,   // [E,3,32,16]
                                float* __restrict__ out)            // [E*G,512]
{
    __shared__ __align__(16) __bf16 s_img[HP * HP * CH];  // 68 KB padded image
    __shared__ float s_xsum[8];      // fp32 per-channel input sums (for skip)
    __shared__ float s_pooled[16];   // relu-branch pooled sums
    __shared__ float s_final[16];    // final pooled vector

    const int wg   = blockIdx.x;     // output row (class-major order)
    const int tid  = threadIdx.x;
    const int e    = wg >> 9;        // expert = wg / 512
    const int samp = indices[wg];

    // ---- phase A: zero halo + reduction slots ----
    if (tid < 16) s_pooled[tid] = 0.0f;
    if (tid < 8)  s_xsum[tid]   = 0.0f;
    for (int hx = tid; hx < 2 * HP + 2 * HW; hx += 256) {   // 260 halo pixels
        int r, c;
        if (hx < HP)               { r = 0;            c = hx;            }
        else if (hx < 2 * HP)      { r = HP - 1;       c = hx - HP;       }
        else if (hx < 2 * HP + HW) { r = hx - 2*HP + 1; c = 0;            }
        else                       { r = hx - (2*HP + HW) + 1; c = HP - 1; }
        *(v8bf*)(s_img + (r * HP + c) * CH) = (v8bf){};
    }
    __syncthreads();

    // ---- phase B: stage image to LDS (bf16) + fp32 channel sums ----
    const float4* gin = (const float4*)(inputs + (size_t)samp * IMG_ELEMS);
    float4 xs = make_float4(0.f, 0.f, 0.f, 0.f);
    for (int i = tid; i < IMG_ELEMS / 4; i += 256) {
        __builtin_prefetch(gin + i + 1024, 0, 0);   // global_prefetch_b8
        float4 f = gin[i];
        xs.x += f.x; xs.y += f.y; xs.z += f.z; xs.w += f.w;
        int p = i >> 1, hf = i & 1;                 // pixel, channel-half
        int off = (((p >> 6) + 1) * HP + (p & 63) + 1) * CH + hf * 4;
        v4bf h;
        h[0] = (__bf16)f.x; h[1] = (__bf16)f.y;
        h[2] = (__bf16)f.z; h[3] = (__bf16)f.w;
        *(v4bf*)(s_img + off) = h;
    }
    {   // stride 256 keeps each thread on a fixed 4-channel group
        int c0 = (tid & 1) * 4;
        atomicAdd(&s_xsum[c0 + 0], xs.x);
        atomicAdd(&s_xsum[c0 + 1], xs.y);
        atomicAdd(&s_xsum[c0 + 2], xs.z);
        atomicAdd(&s_xsum[c0 + 3], xs.w);
    }
    __syncthreads();

    // ---- phase C: fused conv GEMM via WMMA ----
    const int lane = tid & 31;
    const int wave = tid >> 5;
    const int m    = lane & 15;      // pixel within M-tile
    const int half = lane >> 4;      // K sub-block selector

    const __bf16* bf_e = bfrag + (size_t)e * FRAG_PER_E;
    v16bf B0 = *(const v16bf*)(bf_e + (0 * 32 + lane) * 16);
    v16bf B1 = *(const v16bf*)(bf_e + (1 * 32 + lane) * 16);
    v16bf B2 = *(const v16bf*)(bf_e + (2 * 32 + lane) * 16);

    // padded-image element offsets of taps relative to (row, col)*:
    //   tap t -> ((t/3)*HP + t%3) * CH ; all reads in-bounds thanks to halo
    const int ks0_lo = half * CH;                               // taps 0/1
    const int ks0_hi = half ? (1 * HP + 0) * CH : (0 * HP + 2) * CH;  // taps 2/3
    const int ks1_lo = (1 * HP + 1) * CH + half * CH;           // taps 4/5
    const int ks1_hi = (2 * HP + 0) * CH + half * CH;           // taps 6/7
    const v8bf zero = {};

    float psum = 0.0f;               // running sum for channel (lane & 15)
    for (int tt = wave; tt < 256; tt += 8) {
        int p    = tt * 16 + m;
        int base = ((p >> 6) * HP + (p & 63)) * CH;   // padded (row, col)
        v8f acc = {};
        union { v16bf v; v8bf h[2]; } a;

        a.h[0] = *(const v8bf*)(s_img + base + ks0_lo);
        a.h[1] = *(const v8bf*)(s_img + base + ks0_hi);
        acc = __builtin_amdgcn_wmma_f32_16x16x32_bf16(
                  false, a.v, false, B0, (short)0, acc, false, false);

        a.h[0] = *(const v8bf*)(s_img + base + ks1_lo);
        a.h[1] = *(const v8bf*)(s_img + base + ks1_hi);
        acc = __builtin_amdgcn_wmma_f32_16x16x32_bf16(
                  false, a.v, false, B1, (short)0, acc, false, false);

        // tap 8 valid only for lane-half 0; invalid lanes read zeroed halo px 0
        int a2 = half ? 0 : (base + (2 * HP + 2) * CH);
        a.h[0] = *(const v8bf*)(s_img + a2);
        a.h[1] = zero;
        acc = __builtin_amdgcn_wmma_f32_16x16x32_bf16(
                  false, a.v, false, B2, (short)0, acc, false, false);

#pragma unroll
        for (int i = 0; i < 8; ++i)
            psum += fmaxf(acc[i], 0.0f);              // relu + pool
    }

    psum += __shfl_xor(psum, 16, 32);                 // fold lane halves
    if (lane < 16) atomicAdd(&s_pooled[lane], psum);  // ds_add_f32
    __syncthreads();

    // ---- phase D: add analytic fp32 skip term, scale by 1/PIX ----
    if (tid < 16) {
        float acc = s_pooled[tid];
        const float* ws_e = Wskip + (size_t)e * 128;  // [8,16]
#pragma unroll
        for (int c = 0; c < 8; ++c)
            acc = fmaf(s_xsum[c], ws_e[c * 16 + tid], acc);
        s_final[tid] = acc * (1.0f / (float)PIX);
    }
    __syncthreads();

    // ---- phase E: dense head pooled[16] @ Wd[e] * mask[e] ----
    float pl[16];
#pragma unroll
    for (int k = 0; k < 16; ++k) pl[k] = s_final[k];

    const float* wd_e = Wd    + (size_t)e * 16 * MOUT;
    const float* mk_e = cmask + (size_t)e * MOUT;
    for (int o = tid; o < MOUT; o += 256) {
        float acc = 0.0f;
#pragma unroll
        for (int k = 0; k < 16; ++k)
            acc = fmaf(pl[k], wd_e[k * MOUT + o], acc);
        out[(size_t)wg * MOUT + o] = acc * mk_e[o];
    }
}

// ---------------------------------------------------------------------------
extern "C" void kernel_launch(void* const* d_in, const int* in_sizes, int n_in,
                              void* d_out, int out_size, void* d_ws, size_t ws_size,
                              hipStream_t stream)
{
    (void)in_sizes; (void)n_in; (void)out_size; (void)ws_size;

    const float* inputs  = (const float*)d_in[0];
    const int*   indices = (const int*)  d_in[1];
    const float* W1      = (const float*)d_in[2];
    const float* W3      = (const float*)d_in[3];
    const float* Wskip   = (const float*)d_in[4];
    const float* Wd      = (const float*)d_in[5];
    const float* cmask   = (const float*)d_in[6];

    __bf16* frag = (__bf16*)d_ws;   // EXPERTS*FRAG_PER_E*2 = 12 KB scratch

    int prep_elems  = EXPERTS * FRAG_PER_E;
    int prep_blocks = (prep_elems + 255) / 256;
    hipLaunchKernelGGL(prep_bfrag_kernel, dim3(prep_blocks), dim3(256), 0, stream,
                       W1, W3, frag);

    hipLaunchKernelGGL(moe_conv_pool_dense_kernel, dim3(NSAMP), dim3(256), 0, stream,
                       inputs, indices, Wskip, Wd, cmask, frag, (float*)d_out);
}
